// MFB_87067577025372
// MI455X (gfx1250) — matrix operand order repaired and checked by reading, not compiled, not run on hardware
//
#include <hip/hip_runtime.h>
#include <hip/hip_bf16.h>

#define BN 4
#define CC 512
#define ICN 256
#define NN 4096
#define EPSF 1e-5f

typedef __attribute__((ext_vector_type(16))) _Float16 v16h;
typedef __attribute__((ext_vector_type(8)))  _Float16 v8h;
typedef __attribute__((ext_vector_type(8)))  float    v8f;
typedef __attribute__((ext_vector_type(4)))  unsigned int u32x4;
typedef __attribute__((ext_vector_type(8)))  int i32x8;
typedef __attribute__((ext_vector_type(4)))  int i32x4;

// ---------------------------------------------------------------------------
// WMMA helpers
// ---------------------------------------------------------------------------
__device__ __forceinline__ v8f wmma_f16(v16h a, v16h b, v8f c) {
    return __builtin_amdgcn_wmma_f32_16x16x32_f16(false, a, false, b,
                                                  (short)0, c, false, false);
}

// A-operand fragment (16 rows x 32 K, f16), contiguous layout:
// lane<16: K = 0..15 ; lane>=16: K = 16..31 ; row = lane%16
__device__ __forceinline__ v16h load_afrag(const _Float16* base, int row0,
                                           int stride, int k0, int lane) {
    int r  = lane & 15;
    int hi = lane >> 4;
    const _Float16* p = base + (size_t)(row0 + r) * stride + k0 + hi * 16;
    v8h lo = *(const v8h*)p;
    v8h hi8 = *(const v8h*)(p + 8);
    return __builtin_shufflevector(lo, hi8, 0,1,2,3,4,5,6,7,8,9,10,11,12,13,14,15);
}

// B-operand fragment (32 K x 16 cols, f16): lane holds column lane%16 of B,
// K = 0..15 (lane<16) or 16..31 (lane>=16), contiguous from B^T row.
__device__ __forceinline__ v16h load_bfrag(const _Float16* base, int col0,
                                           int stride, int k0, int lane) {
    int r  = lane & 15;
    int hi = lane >> 4;
    const _Float16* p = base + (size_t)(col0 + r) * stride + k0 + hi * 16;
    return *(const v16h*)p;
}

// ---------------------------------------------------------------------------
// Tensor Data Mover: async 2-D tile load (global f16 tensor -> LDS), per
// CDNA5 ISA S#8.  D# group0/group1 built from literals +
// readfirstlane'd address words so they land in SGPRs.
// ---------------------------------------------------------------------------
__device__ __forceinline__ void tdm_load_2d(unsigned int lds_off, unsigned long long gaddr,
                                            unsigned int tile_w, unsigned int tile_h,
                                            unsigned int row_stride,
                                            unsigned int tensor_w, unsigned int tensor_h) {
    unsigned int glo = (unsigned int)__builtin_amdgcn_readfirstlane((int)(gaddr & 0xffffffffu));
    unsigned int ghi = (unsigned int)__builtin_amdgcn_readfirstlane((int)(gaddr >> 32));
    unsigned int loff = (unsigned int)__builtin_amdgcn_readfirstlane((int)lds_off);
    u32x4 g0;
    g0[0] = 1u;                                  // count=1, user mode, no gather
    g0[1] = loff;                                // lds_addr
    g0[2] = glo;                                 // global_addr[31:0]
    g0[3] = (ghi & 0x01ffffffu) | 0x80000000u;   // global_addr[56:32] | type=2
    i32x8 g1;
    g1[0] = (int)(1u << 16);                     // wg_mask=0, data_size=1 (2 bytes)
    g1[1] = (int)((tensor_w & 0xffffu) << 16);   // tensor_dim0[15:0]
    g1[2] = (int)((tensor_w >> 16) | ((tensor_h & 0xffffu) << 16)); // dim0 hi | dim1 lo
    g1[3] = (int)((tensor_h >> 16) | (tile_w << 16));               // dim1 hi | tile_dim0
    g1[4] = (int)tile_h;                         // tile_dim1 (tile_dim2 = 0)
    g1[5] = (int)row_stride;                     // tensor_dim0_stride[31:0]
    g1[6] = 0;
    g1[7] = 0;
    i32x4 z4 = {0, 0, 0, 0};
#if defined(__clang_major__) && (__clang_major__ >= 23)
    i32x8 z8 = {0, 0, 0, 0, 0, 0, 0, 0};
    __builtin_amdgcn_tensor_load_to_lds(g0, g1, z4, z4, z8, 0);
#else
    __builtin_amdgcn_tensor_load_to_lds(g0, g1, z4, z4, 0);
#endif
}

__device__ __forceinline__ unsigned int lds_byte_off(const void* p) {
    // low 32 bits of a flat LDS pointer == byte offset within the WG allocation
    return (unsigned int)(unsigned long long)(uintptr_t)p;
}

// ---------------------------------------------------------------------------
// Kernel 1: convert weights f32 -> f16 copies in workspace
// ---------------------------------------------------------------------------
__global__ void cvt_weights(const float* wq, const float* wk, const float* wv,
                            const float* wwg, const float* wout,
                            _Float16* q16, _Float16* k16, _Float16* v16,
                            _Float16* wg16, _Float16* wo16) {
    int i = blockIdx.x * blockDim.x + threadIdx.x;
    if (i < ICN * CC) {
        q16[i] = (_Float16)wq[i];
        k16[i] = (_Float16)wk[i];
        v16[i] = (_Float16)wv[i];
        wo16[i] = (_Float16)wout[i];
    }
    if (i < ICN * ICN) wg16[i] = (_Float16)wwg[i];
}

// ---------------------------------------------------------------------------
// Kernel 2: transpose xA [B,C,N] f32 -> Xt [B,N,C] f16 via LDS 32x32 tiles
// ---------------------------------------------------------------------------
__global__ __launch_bounds__(256) void transpose_x(const float* __restrict__ x,
                                                   _Float16* __restrict__ xt) {
    __shared__ _Float16 t[32][33];
    int n0 = blockIdx.x * 32, c0 = blockIdx.y * 32, b = blockIdx.z;
    const float* xb = x + (size_t)b * CC * NN;
    _Float16* xtb = xt + (size_t)b * NN * CC;
    int tx = threadIdx.x, ty = threadIdx.y;   // 32 x 8
#pragma unroll
    for (int i = 0; i < 32; i += 8)
        t[ty + i][tx] = (_Float16)xb[(size_t)(c0 + ty + i) * NN + n0 + tx];
    __syncthreads();
#pragma unroll
    for (int i = 0; i < 32; i += 8)
        xtb[(size_t)(n0 + ty + i) * CC + c0 + tx] = t[tx][ty + i];
}

// ---------------------------------------------------------------------------
// Kernel 3: fused QKV GEMM.  Y = W @ X + b for q,k,v sharing the X fragments.
// ---------------------------------------------------------------------------
__global__ __launch_bounds__(256) void qkv_gemm(
    const _Float16* __restrict__ xt,
    const _Float16* __restrict__ wq16, const _Float16* __restrict__ wk16,
    const _Float16* __restrict__ wv16,
    const float* __restrict__ bq, const float* __restrict__ bk,
    const float* __restrict__ bv,
    _Float16* __restrict__ Qt, _Float16* __restrict__ Vt,
    _Float16* __restrict__ Kc) {
    int lane = threadIdx.x & 31, wave = threadIdx.x >> 5;
    int tile = blockIdx.x * 8 + wave;          // B * 256 * 16 tiles
    int b = tile >> 12;
    int rem = tile & 4095;
    int tok0 = (rem >> 4) * 16, ic0 = (rem & 15) * 16;
    const _Float16* xtb = xt + (size_t)b * NN * CC;

    v8f aq = {}, ak = {}, av = {};
#pragma unroll
    for (int kc = 0; kc < CC; kc += 32) {
        v16h bf = load_bfrag(xtb, tok0, CC, kc, lane);
        aq = wmma_f16(load_afrag(wq16, ic0, CC, kc, lane), bf, aq);
        ak = wmma_f16(load_afrag(wk16, ic0, CC, kc, lane), bf, ak);
        av = wmma_f16(load_afrag(wv16, ic0, CC, kc, lane), bf, av);
    }
    int r16 = lane & 15, hi = lane >> 4;
    v8h qh, vh;
#pragma unroll
    for (int r = 0; r < 8; r++) {
        int ic = ic0 + r + 8 * hi;
        qh[r] = (_Float16)(aq[r] + bq[ic]);
        vh[r] = (_Float16)(av[r] + bv[ic]);
    }
    size_t rowoff = (size_t)b * NN * ICN + (size_t)(tok0 + r16) * ICN + ic0 + hi * 8;
    *(v8h*)(Qt + rowoff) = qh;
    *(v8h*)(Vt + rowoff) = vh;
    _Float16* kb = Kc + (size_t)b * ICN * NN;
#pragma unroll
    for (int r = 0; r < 8; r++) {
        int ic = ic0 + r + 8 * hi;
        kb[(size_t)ic * NN + tok0 + r16] = (_Float16)(ak[r] + bk[ic]);
    }
}

// ---------------------------------------------------------------------------
// Kernel 4 (pass A): per key-row n, online max/sum of exp over all m of
// S[n,m].  Computed as S^T tiles [m x n] so the reduction is column-wise.
// ---------------------------------------------------------------------------
__global__ __launch_bounds__(128) void attn_stats(
    const _Float16* __restrict__ Qt, const _Float16* __restrict__ Vt,
    float* __restrict__ Mst, float* __restrict__ Lst) {
    int lane = threadIdx.x & 31, wave = threadIdx.x >> 5;
    int b = blockIdx.x >> 6;                       // 256 blocks -> B
    int n0 = (((blockIdx.x & 63) << 2) + wave) * 16;
    const _Float16* qb = Qt + (size_t)b * NN * ICN;
    const _Float16* vb = Vt + (size_t)b * NN * ICN;

    v16h bf[8];
#pragma unroll
    for (int c = 0; c < 8; c++) bf[c] = load_bfrag(vb, n0, ICN, c * 32, lane);

    float M = -1e30f, L = 0.f;
    for (int m0 = 0; m0 < NN; m0 += 16) {
        v8f d = {};
#pragma unroll
        for (int c = 0; c < 8; c++)
            d = wmma_f16(load_afrag(qb, m0, ICN, c * 32, lane), bf[c], d);
        float tmax = d[0];
#pragma unroll
        for (int r = 1; r < 8; r++) tmax = fmaxf(tmax, d[r]);
        tmax = fmaxf(tmax, __shfl_xor(tmax, 16, 32));
        float newM = fmaxf(M, tmax);
        float part = 0.f;
#pragma unroll
        for (int r = 0; r < 8; r++) part += __expf(d[r] - newM);
        part += __shfl_xor(part, 16, 32);
        L = L * __expf(M - newM) + part;
        M = newM;
    }
    if (lane < 16) {
        Mst[b * NN + n0 + lane] = M;
        Lst[b * NN + n0 + lane] = L;
    }
}

// ---------------------------------------------------------------------------
// Kernel 5 (pass B): av[c,m] = sum_n k[c,n] * exp(S[n,m]-M_n)/L_n.
// TDM double-buffers the shared V/K key slabs into LDS (one issue per block,
// s_wait_tensorcnt + barriers); all 4 waves read fragments from LDS.
// ---------------------------------------------------------------------------
__global__ __launch_bounds__(128) void attn_av(
    const _Float16* __restrict__ Qt, const _Float16* __restrict__ Vt,
    const _Float16* __restrict__ Kc,
    const float* __restrict__ Mst, const float* __restrict__ Lst,
    _Float16* __restrict__ avT) {
    __shared__ _Float16 ldsV[2][32 * ICN];     // 2 x 16KB: V rows n0..n0+31
    __shared__ _Float16 ldsK[2][ICN * 32];     // 2 x 16KB: K tile [256c x 32n]
    int lane = threadIdx.x & 31, wave = threadIdx.x >> 5;
    int b = blockIdx.x >> 6;                       // uniform (blockIdx only)
    int m0 = (((blockIdx.x & 63) << 2) + wave) * 16;
    const _Float16* qb = Qt + (size_t)b * NN * ICN;
    const _Float16* vb = Vt + (size_t)b * NN * ICN;
    const _Float16* kb = Kc + (size_t)b * ICN * NN;
    const float* Mb = Mst + b * NN;
    const float* Lb = Lst + b * NN;
    int r16 = lane & 15, hi = lane >> 4;

    v16h qf[8];
#pragma unroll
    for (int c = 0; c < 8; c++) qf[c] = load_bfrag(qb, m0, ICN, c * 32, lane);

    v8f acc[16];
#pragma unroll
    for (int ct = 0; ct < 16; ct++) acc[ct] = (v8f){};

    // prologue: stage chunk 0 into buffer 0
    if (wave == 0) {
        tdm_load_2d(lds_byte_off(&ldsV[0][0]), (unsigned long long)(uintptr_t)vb,
                    ICN, 32, ICN, ICN, NN);
        tdm_load_2d(lds_byte_off(&ldsK[0][0]), (unsigned long long)(uintptr_t)kb,
                    32, ICN, NN, NN, ICN);
    }

    for (int n0 = 0; n0 < NN; n0 += 32) {
        int buf = (n0 >> 5) & 1;
        if (wave == 0) {
            if (n0 + 32 < NN) {
                tdm_load_2d(lds_byte_off(&ldsV[buf ^ 1][0]),
                            (unsigned long long)(uintptr_t)(vb + (size_t)(n0 + 32) * ICN),
                            ICN, 32, ICN, ICN, NN);
                tdm_load_2d(lds_byte_off(&ldsK[buf ^ 1][0]),
                            (unsigned long long)(uintptr_t)(kb + n0 + 32),
                            32, ICN, NN, NN, ICN);
                __builtin_amdgcn_s_wait_tensorcnt(2);   // chunk n0 complete
            } else {
                __builtin_amdgcn_s_wait_tensorcnt(0);
            }
        }
        __syncthreads();                               // slab ready for all waves

        const _Float16* vs = &ldsV[buf][0];            // stride ICN, rows 0..31
        const _Float16* ks = &ldsK[buf][0];            // stride 32, rows = c
        // S tiles: rows = keys n, cols = m
        v8f s0 = {}, s1 = {};
#pragma unroll
        for (int c = 0; c < 8; c++) {
            s0 = wmma_f16(load_afrag(vs, 0,  ICN, c * 32, lane), qf[c], s0);
            s1 = wmma_f16(load_afrag(vs, 16, ICN, c * 32, lane), qf[c], s1);
        }
        // P = exp(S - M_n) / L_n   (row-indexed scale)
        v8h h0, h1;
#pragma unroll
        for (int r = 0; r < 8; r++) {
            int na = n0 + r + 8 * hi;
            h0[r] = (_Float16)(__expf(s0[r] - Mb[na]) * (1.0f / Lb[na]));
            int nb = na + 16;
            h1[r] = (_Float16)(__expf(s1[r] - Mb[nb]) * (1.0f / Lb[nb]));
        }
        // Repack D layout -> B layout via one shfl_xor(16) exchange
        union { v8h h; int i[4]; } snd, rcv;
        snd.h = hi ? h0 : h1;
#pragma unroll
        for (int j = 0; j < 4; j++) rcv.i[j] = __shfl_xor(snd.i[j], 16, 32);
        v8h plo = hi ? rcv.h : h0;
        v8h phi = hi ? h1 : rcv.h;
        v16h pf = __builtin_shufflevector(plo, phi,
                    0,1,2,3,4,5,6,7,8,9,10,11,12,13,14,15);
        // av += K_tile(16c x 32n) * P(32n x 16m)
#pragma unroll
        for (int ct = 0; ct < 16; ct++)
            acc[ct] = wmma_f16(load_afrag(ks, ct * 16, 32, 0, lane), pf, acc[ct]);
        __syncthreads();                               // done reading this slab
    }
    _Float16* arow = avT + (size_t)b * NN * ICN + (size_t)(m0 + r16) * ICN + hi * 8;
#pragma unroll
    for (int ct = 0; ct < 16; ct++) {
        v8h h;
#pragma unroll
        for (int r = 0; r < 8; r++) h[r] = (_Float16)acc[ct][r];
        *(v8h*)(arow + ct * 16) = h;
    }
}

// ---------------------------------------------------------------------------
// Kernel 6: avw = wwg @ av, then BatchNorm1d (eval).  avT/avwT token-major.
// ---------------------------------------------------------------------------
__global__ __launch_bounds__(256) void avw_gemm(
    const _Float16* __restrict__ avT, const _Float16* __restrict__ wg16,
    const float* __restrict__ g, const float* __restrict__ bt,
    const float* __restrict__ mn, const float* __restrict__ vr,
    _Float16* __restrict__ avwT) {
    int lane = threadIdx.x & 31, wave = threadIdx.x >> 5;
    int tile = blockIdx.x * 8 + wave;
    int b = tile >> 12;
    int rem = tile & 4095;
    int tok0 = (rem >> 4) * 16, oc0 = (rem & 15) * 16;
    const _Float16* ab = avT + (size_t)b * NN * ICN;

    v8f acc = {};
#pragma unroll
    for (int kc = 0; kc < ICN; kc += 32) {
        v16h bf = load_bfrag(ab, tok0, ICN, kc, lane);
        acc = wmma_f16(load_afrag(wg16, oc0, ICN, kc, lane), bf, acc);
    }
    int r16 = lane & 15, hi = lane >> 4;
    v8h h;
#pragma unroll
    for (int r = 0; r < 8; r++) {
        int oc = oc0 + r + 8 * hi;
        float s = g[oc] * rsqrtf(vr[oc] + EPSF);
        h[r] = (_Float16)((acc[r] - mn[oc]) * s + bt[oc]);
    }
    *(v8h*)(avwT + (size_t)b * NN * ICN + (size_t)(tok0 + r16) * ICN + oc0 + hi * 8) = h;
}

// ---------------------------------------------------------------------------
// Kernel 7: o = wout @ avw + bout, BatchNorm2d (eval), +residual, ReLU -> f32
// ---------------------------------------------------------------------------
__global__ __launch_bounds__(256) void out_gemm(
    const _Float16* __restrict__ avwT, const _Float16* __restrict__ wo16,
    const float* __restrict__ bout,
    const float* __restrict__ g2, const float* __restrict__ b2,
    const float* __restrict__ m2, const float* __restrict__ v2,
    const float* __restrict__ xA, float* __restrict__ out) {
    int lane = threadIdx.x & 31, wave = threadIdx.x >> 5;
    int tile = blockIdx.x * 8 + wave;          // B * 256 * 32 tiles
    int b = tile >> 13;
    int rem = tile & 8191;
    int tok0 = (rem >> 5) * 16, c0 = (rem & 31) * 16;
    const _Float16* ab = avwT + (size_t)b * NN * ICN;

    v8f acc = {};
#pragma unroll
    for (int kc = 0; kc < ICN; kc += 32) {
        v16h bf = load_bfrag(ab, tok0, ICN, kc, lane);
        acc = wmma_f16(load_afrag(wo16, c0, ICN, kc, lane), bf, acc);
    }
    int r16 = lane & 15, hi = lane >> 4;
    const float* xb = xA + (size_t)b * CC * NN;
    float* ob = out + (size_t)b * CC * NN;
#pragma unroll
    for (int r = 0; r < 8; r++) {
        int c = c0 + r + 8 * hi;
        size_t idx = (size_t)c * NN + tok0 + r16;
        float s = g2[c] * rsqrtf(v2[c] + EPSF);
        float y = (acc[r] + bout[c] - m2[c]) * s + b2[c] + xb[idx];
        ob[idx] = fmaxf(y, 0.f);
    }
}

// ---------------------------------------------------------------------------
// Host launcher
// ---------------------------------------------------------------------------
extern "C" void kernel_launch(void* const* d_in, const int* in_sizes, int n_in,
                              void* d_out, int out_size, void* d_ws, size_t ws_size,
                              hipStream_t stream) {
    const float* xA   = (const float*)d_in[0];
    const float* wk   = (const float*)d_in[1];
    const float* bk   = (const float*)d_in[2];
    const float* wv   = (const float*)d_in[3];
    const float* bv   = (const float*)d_in[4];
    const float* wq   = (const float*)d_in[5];
    const float* bq   = (const float*)d_in[6];
    const float* wwg  = (const float*)d_in[7];
    const float* bn1g = (const float*)d_in[8];
    const float* bn1b = (const float*)d_in[9];
    const float* bn1m = (const float*)d_in[10];
    const float* bn1v = (const float*)d_in[11];
    const float* wout = (const float*)d_in[12];
    const float* bout = (const float*)d_in[13];
    const float* bn2g = (const float*)d_in[14];
    const float* bn2b = (const float*)d_in[15];
    const float* bn2m = (const float*)d_in[16];
    const float* bn2v = (const float*)d_in[17];
    float* out = (float*)d_out;

    char* ws = (char*)d_ws;
    size_t off = 0;
    auto take = [&](size_t bytes) {
        char* p = ws + off;
        off += (bytes + 255) & ~(size_t)255;
        return p;
    };
    _Float16* wq16  = (_Float16*)take((size_t)ICN * CC * 2);
    _Float16* wk16  = (_Float16*)take((size_t)ICN * CC * 2);
    _Float16* wv16  = (_Float16*)take((size_t)ICN * CC * 2);
    _Float16* wg16  = (_Float16*)take((size_t)ICN * ICN * 2);
    _Float16* wo16  = (_Float16*)take((size_t)CC * ICN * 2);
    _Float16* Xt    = (_Float16*)take((size_t)BN * NN * CC * 2);
    _Float16* Qt    = (_Float16*)take((size_t)BN * NN * ICN * 2);
    _Float16* Vt    = (_Float16*)take((size_t)BN * NN * ICN * 2);
    _Float16* Kc    = (_Float16*)take((size_t)BN * NN * ICN * 2);
    float*    Mst   = (float*)take((size_t)BN * NN * 4);
    float*    Lst   = (float*)take((size_t)BN * NN * 4);
    _Float16* avT   = (_Float16*)take((size_t)BN * NN * ICN * 2);
    _Float16* avwT  = (_Float16*)take((size_t)BN * NN * ICN * 2);
    (void)ws_size; (void)in_sizes; (void)n_in; (void)out_size;

    cvt_weights<<<512, 256, 0, stream>>>(wq, wk, wv, wwg, wout,
                                         wq16, wk16, wv16, wg16, wo16);
    transpose_x<<<dim3(NN / 32, CC / 32, BN), dim3(32, 8), 0, stream>>>(xA, Xt);
    qkv_gemm<<<2048, 256, 0, stream>>>(Xt, wq16, wk16, wv16, bq, bk, bv,
                                       Qt, Vt, Kc);
    attn_stats<<<256, 128, 0, stream>>>(Qt, Vt, Mst, Lst);
    attn_av<<<256, 128, 0, stream>>>(Qt, Vt, Kc, Mst, Lst, avT);
    avw_gemm<<<2048, 256, 0, stream>>>(avT, wg16, bn1g, bn1b, bn1m, bn1v, avwT);
    out_gemm<<<4096, 256, 0, stream>>>(avwT, wo16, bout, bn2g, bn2b, bn2m, bn2v,
                                       xA, out);
}
